// Attention_18811956756912
// MI455X (gfx1250) — matrix operand order repaired and checked by reading, not compiled, not run on hardware
//
#include <hip/hip_runtime.h>

// ---------------------------------------------------------------------------
// bf16 WMMA plumbing (CDNA5 / gfx1250, wave32)
// ---------------------------------------------------------------------------
typedef __attribute__((ext_vector_type(16))) __bf16 v16bf;
typedef __attribute__((ext_vector_type(8)))  float  v8f;
typedef __attribute__((ext_vector_type(4))) unsigned int u32x4;
typedef __attribute__((ext_vector_type(8)))  int  i32x8;
typedef __attribute__((ext_vector_type(4)))  int  i32x4;

union Frag16 { v16bf v; unsigned int u[8]; };
union U4     { uint4 q; unsigned int ui[4]; unsigned short us[8]; };

// Probe-confirmed CDNA5 TDM builtin (arity differs by toolchain).
#if __has_builtin(__builtin_amdgcn_tensor_load_to_lds)
#define HAVE_TDM 1
#else
#define HAVE_TDM 0
#endif

__device__ __forceinline__ void wait_tensor1() {
#if __has_builtin(__builtin_amdgcn_s_wait_tensorcnt)
  __builtin_amdgcn_s_wait_tensorcnt(1);
#else
  asm volatile("s_wait_tensorcnt 0x1" ::: "memory");
#endif
}
__device__ __forceinline__ void wait_tensor0() {
#if __has_builtin(__builtin_amdgcn_s_wait_tensorcnt)
  __builtin_amdgcn_s_wait_tensorcnt(0);
#else
  asm volatile("s_wait_tensorcnt 0x0" ::: "memory");
#endif
}

__device__ __forceinline__ unsigned short f2bf(float f) {
  unsigned int u = __float_as_uint(f);
  return (unsigned short)((u + 0x7FFFu + ((u >> 16) & 1u)) >> 16);  // RNE
}
__device__ __forceinline__ unsigned int pack2bf(float a, float b) {
  return (unsigned int)f2bf(a) | ((unsigned int)f2bf(b) << 16);
}
__device__ __forceinline__ v8f wmma_bf16(Frag16 a, Frag16 b, v8f c) {
  return __builtin_amdgcn_wmma_f32_16x16x32_bf16(
      /*neg_a=*/false, a.v, /*neg_b=*/false, b.v,
      /*c_mod=*/(short)0, c, /*reuse_a=*/false, /*reuse_b=*/false);
}

#if HAVE_TDM
// 2D TDM descriptor: load tile [32 rows][1024 bf16], row stride 1024 elems,
// from global tile-start gsrc into LDS byte offset lds_off.
// Bit layout per CDNA5 ISA sect. 8.3/8.4.
__device__ __forceinline__ void tdm_load_tile_k(unsigned lds_off,
                                                const unsigned short* gsrc) {
  unsigned long long ga = (unsigned long long)(uintptr_t)gsrc;
  u32x4 g0;
  g0[0] = 1u;                                   // count=1, user descriptor
  g0[1] = lds_off;                              // lds_addr [63:32]
  g0[2] = (unsigned)ga;                         // global_addr [95:64]
  g0[3] = (unsigned)((ga >> 32) & 0x01FFFFFFull) | (2u << 30);  // addr hi | type=2
  i32x8 g1;
  g1[0] = 1 << 16;            // [17:16] data_size = 1 (2 bytes)
  g1[1] = 0x0400 << 16;       // tensor_dim0 = 1024 (low 16 -> [63:48])
  g1[2] = 0x1000 << 16;       // tensor_dim0 hi=0; tensor_dim1 = 4096 (low 16)
  g1[3] = 0x0400 << 16;       // tensor_dim1 hi=0; tile_dim0 = 1024
  g1[4] = 32;                 // tile_dim1 = 32; tile_dim2 = 0
  g1[5] = 1024;               // tensor_dim0_stride = 1024 elements
  g1[6] = 0;                  // stride hi, tensor_dim1_stride lo
  g1[7] = 0;
  i32x4 gz = i32x4{0, 0, 0, 0};
#if __clang_major__ >= 23
  i32x8 gz8 = i32x8{0, 0, 0, 0, 0, 0, 0, 0};
  __builtin_amdgcn_tensor_load_to_lds(g0, g1, gz, gz, gz8, 0);
#else
  __builtin_amdgcn_tensor_load_to_lds(g0, g1, gz, gz, 0);
#endif
}
#endif

// ---------------------------------------------------------------------------
// Projections: C[M,N](bf16) = scale * A[M,K=1024](f32) @ W[N,K=1024](f32)^T
// VT==1: write transposed into vT[b][n][j] (b = m>>11, j = m&2047) with a
//        single b128 store per 16x16 tile (lane's 8 rows = 8 consecutive j).
// ---------------------------------------------------------------------------
template <int VT>
__global__ __launch_bounds__(256, 2)
void gemm_xWT_bf16(const float* __restrict__ A, const float* __restrict__ W,
                   unsigned short* __restrict__ out, int M, int N, float scale)
{
  const int K = 1024;
  __shared__ unsigned short a_sh[128 * 32];
  __shared__ unsigned short b_sh[128 * 32];

  const int t    = threadIdx.x;
  const int lane = t & 31;
  const int w    = t >> 5;
  const int wm   = w >> 1;
  const int wn   = w & 1;
  const int tilesM = M >> 7;
  const int m0 = (blockIdx.x % tilesM) << 7;
  const int n0 = (blockIdx.x / tilesM) << 7;

  v8f acc[2][4];
#pragma unroll
  for (int a = 0; a < 2; ++a)
#pragma unroll
    for (int b = 0; b < 4; ++b)
      acc[a][b] = v8f{0.f,0.f,0.f,0.f,0.f,0.f,0.f,0.f};

  const int r = t >> 1, half = t & 1;
  const float* ap = A + (size_t)(m0 + r) * K + half * 16;
  const float* bp = W + (size_t)(n0 + r) * K + half * 16;
  unsigned int* a_shu = (unsigned int*)a_sh;
  unsigned int* b_shu = (unsigned int*)b_sh;
  const int shbase = (r * 32 + half * 16) >> 1;

  const int arow = lane & 15;
  const int kb   = (lane >> 4) * 8;
  const int kb2  = (lane >> 4) * 16;

  for (int k0 = 0; k0 < K; k0 += 32) {
    __syncthreads();
    __builtin_prefetch(ap + k0 + 32, 0, 1);
    __builtin_prefetch(bp + k0 + 32, 0, 1);
#pragma unroll
    for (int g = 0; g < 4; ++g) {
      float4 av = *(const float4*)(ap + k0 + g * 4);
      float4 bv = *(const float4*)(bp + k0 + g * 4);
      a_shu[shbase + g * 2 + 0] = pack2bf(av.x, av.y);
      a_shu[shbase + g * 2 + 1] = pack2bf(av.z, av.w);
      b_shu[shbase + g * 2 + 0] = pack2bf(bv.x, bv.y);
      b_shu[shbase + g * 2 + 1] = pack2bf(bv.z, bv.w);
    }
    __syncthreads();

    Frag16 afr[2], bfr[4];
#pragma unroll
    for (int mi = 0; mi < 2; ++mi) {
      const unsigned short* base = a_sh + (wm * 32 + mi * 16 + arow) * 32 + kb;
      U4 lo, hi;
      lo.q = *(const uint4*)(base);
      hi.q = *(const uint4*)(base + 16);
#pragma unroll
      for (int e = 0; e < 4; ++e) { afr[mi].u[e] = lo.ui[e]; afr[mi].u[4 + e] = hi.ui[e]; }
    }
#pragma unroll
    for (int ni = 0; ni < 4; ++ni) {
      const unsigned short* base = b_sh + (wn * 64 + ni * 16 + arow) * 32 + kb;
      U4 lo, hi;
      lo.q = *(const uint4*)(base);
      hi.q = *(const uint4*)(base + 16);
#pragma unroll
      for (int e = 0; e < 4; ++e) { bfr[ni].u[e] = lo.ui[e]; bfr[ni].u[4 + e] = hi.ui[e]; }
    }
#pragma unroll
    for (int mi = 0; mi < 2; ++mi)
#pragma unroll
      for (int ni = 0; ni < 4; ++ni)
        acc[mi][ni] = wmma_bf16(afr[mi], bfr[ni], acc[mi][ni]);
  }

  const int erow = (lane >> 4) * 8, ecol = lane & 15;
#pragma unroll
  for (int mi = 0; mi < 2; ++mi)
#pragma unroll
    for (int ni = 0; ni < 4; ++ni) {
      const int mbase = m0 + wm * 32 + mi * 16 + erow;   // rr adds 0..7
      const int n     = n0 + wn * 64 + ni * 16 + ecol;
      if (VT == 0) {
#pragma unroll
        for (int rr = 0; rr < 8; ++rr)
          out[(size_t)(mbase + rr) * N + n] = f2bf(acc[mi][ni][rr] * scale);
      } else {
        // vT[b][n][j]: lane's 8 rows are 8 consecutive j -> one b128 store.
        const int bb = mbase >> 11, j = mbase & 2047;
        U4 pk;
#pragma unroll
        for (int e = 0; e < 4; ++e)
          pk.ui[e] = pack2bf(acc[mi][ni][2 * e] * scale,
                             acc[mi][ni][2 * e + 1] * scale);
        *(uint4*)(out + (size_t)bb * 1024 * 2048 + (size_t)n * 2048 + j) = pk.q;
      }
    }
}

// ---------------------------------------------------------------------------
// Fused talking-heads attention.
// 16 waves = 16 heads; K tiles via TDM double-buffer; V read from global vT.
// ---------------------------------------------------------------------------
static constexpr unsigned OFF_K    = 0;                    // 2 x [32][1024] u16 = 128KB
static constexpr unsigned OFF_SC   = 131072;               // f32 [16][16][32]  32KB
static constexpr unsigned OFF_AWB  = 131072 + 32768;       // u16 [16][16][32]  16KB
static constexpr unsigned OFF_STM  = OFF_AWB + 16384;      // f32 [256]
static constexpr unsigned OFF_STI  = OFF_STM + 1024;       // f32 [256]
static constexpr unsigned OFF_WT   = OFF_STI + 1024;       // f32 [256]
static constexpr unsigned OFF_G    = OFF_WT + 1024;        // f32 [16]
static constexpr unsigned OFF_B    = OFF_G + 64;           // f32 [16]
static constexpr unsigned LDS_SIZE = OFF_B + 64;           // ~180KB < 320KB/WGP

// Score tiles for one head: S[16 i][32 j] from q-frags (regs) x k tile
// (row-major [32 j][1024 d], stride 1024) -> sc[h][i][j] in LDS.
__device__ __forceinline__ void score_tiles(const unsigned short* ktile,
                                            const Frag16 (&qa)[2],
                                            int h, int lane, float* sc)
{
  const int arow = lane & 15;
  const int kb2  = (lane >> 4) * 8;
  const int mih  = (lane >> 4) * 8;
#pragma unroll
  for (int jsub = 0; jsub < 2; ++jsub) {
    v8f d = v8f{0.f,0.f,0.f,0.f,0.f,0.f,0.f,0.f};
#pragma unroll
    for (int c = 0; c < 2; ++c) {
      Frag16 bfr;
      const unsigned short* base =
          ktile + (size_t)(jsub * 16 + arow) * 1024 + h * 64 + c * 32 + kb2;
      U4 lo, hi;
      lo.q = *(const uint4*)(base);
      hi.q = *(const uint4*)(base + 16);
#pragma unroll
      for (int e = 0; e < 4; ++e) { bfr.u[e] = lo.ui[e]; bfr.u[4 + e] = hi.ui[e]; }
      d = wmma_bf16(qa[c], bfr, d);
    }
#pragma unroll
    for (int rr = 0; rr < 8; ++rr)
      sc[h * 512 + (rr + mih) * 32 + jsub * 16 + arow] = d[rr];
  }
}

__global__ __launch_bounds__(512, 1)
void fused_talking_attn(const unsigned short* __restrict__ qb,
                        const unsigned short* __restrict__ kb,
                        const unsigned short* __restrict__ vtb,
                        const float* __restrict__ Wtalk,
                        const float* __restrict__ gamma,
                        const float* __restrict__ beta,
                        float* __restrict__ out)
{
  extern __shared__ char smem[];
  float*          sc    = (float*)(smem + OFF_SC);
  unsigned short* awb   = (unsigned short*)(smem + OFF_AWB);
  float*          st_m  = (float*)(smem + OFF_STM);
  float*          st_i  = (float*)(smem + OFF_STI);
  float*          wt    = (float*)(smem + OFF_WT);
  float*          g_sh  = (float*)(smem + OFF_G);
  float*          b_sh  = (float*)(smem + OFF_B);

  const int t    = threadIdx.x;
  const int lane = t & 31;
  const int h    = t >> 5;                    // wave index == head
  const int i0   = blockIdx.x * 16;           // flat query row (b*2048 + i)
  const int b    = i0 >> 11;
  const size_t j0base = (size_t)b * 2048;

  if (t < 256) wt[t] = Wtalk[t];
  if (t < 16)  { g_sh[t] = gamma[t]; b_sh[t] = beta[t]; }

  // q A-fragments (16 rows x 64 d) in registers, one head per wave.
  const int arow = lane & 15;
  const int kb_a = (lane >> 4) * 8;
  Frag16 qa[2];
#pragma unroll
  for (int c = 0; c < 2; ++c) {
    const unsigned short* base =
        qb + (size_t)(i0 + arow) * 1024 + h * 64 + c * 32 + kb_a;
    U4 lo, hi;
    lo.q = *(const uint4*)(base);
    hi.q = *(const uint4*)(base + 16);
#pragma unroll
    for (int e = 0; e < 4; ++e) { qa[c].u[e] = lo.ui[e]; qa[c].u[4 + e] = hi.ui[e]; }
  }

#if HAVE_TDM
  const unsigned ldsK = __builtin_amdgcn_groupstaticsize() + OFF_K;
#endif

  // ---- Pass 1: online softmax stats per (head, row) ----
#if HAVE_TDM
  if (h == 0) tdm_load_tile_k(ldsK, kb + j0base * 1024);
#endif
  float m_run = -1e30f, s_run = 0.f;
  for (int j0 = 0; j0 < 2048; j0 += 32) {
    const int buf = (j0 >> 5) & 1;
#if HAVE_TDM
    if (h == 0) {
      if (j0 + 32 < 2048) {
        tdm_load_tile_k(ldsK + (buf ^ 1) * 65536,
                        kb + (j0base + j0 + 32) * 1024);
        wait_tensor1();        // current tile (older transfer) complete
      } else {
        wait_tensor0();
      }
    }
    __syncthreads();
    const unsigned short* ksrc = (const unsigned short*)(smem + OFF_K + buf * 65536);
#else
    __syncthreads();
    const unsigned short* ksrc = kb + (j0base + j0) * 1024;
    (void)buf;
#endif
    score_tiles(ksrc, qa, h, lane, sc);
    __syncthreads();
    if (t < 256) {                            // t = h*16 + i
      const float* row = sc + t * 32;
      float tm = row[0];
#pragma unroll 8
      for (int j = 1; j < 32; ++j) tm = fmaxf(tm, row[j]);
      float m2 = fmaxf(m_run, tm);
      float sa = 0.f;
#pragma unroll 8
      for (int j = 0; j < 32; ++j) sa += __expf(row[j] - m2);
      s_run = s_run * __expf(m_run - m2) + sa;
      m_run = m2;
    }
  }
  __syncthreads();
  if (t < 256) { st_m[t] = m_run; st_i[t] = 1.f / s_run; }

  // ---- Pass 2: recompute scores, talk + LN across heads, PV ----
  v8f acc[4];
#pragma unroll
  for (int c = 0; c < 4; ++c) acc[c] = v8f{0.f,0.f,0.f,0.f,0.f,0.f,0.f,0.f};

#if HAVE_TDM
  if (h == 0) tdm_load_tile_k(ldsK, kb + j0base * 1024);
#endif
  for (int j0 = 0; j0 < 2048; j0 += 32) {
    const int buf = (j0 >> 5) & 1;
#if HAVE_TDM
    if (h == 0) {
      if (j0 + 32 < 2048) {
        tdm_load_tile_k(ldsK + (buf ^ 1) * 65536,
                        kb + (j0base + j0 + 32) * 1024);
        wait_tensor1();
      } else {
        wait_tensor0();
      }
    }
    __syncthreads();
    const unsigned short* ksrc = (const unsigned short*)(smem + OFF_K + buf * 65536);
#else
    __syncthreads();
    const unsigned short* ksrc = kb + (j0base + j0) * 1024;
    (void)buf;
#endif
    score_tiles(ksrc, qa, h, lane, sc);
    __syncthreads();

    // Element phase: one thread per (i,j) of the 16x32 tile.
    {
      const int ii = t >> 5, jj = t & 31;
      float p[16];
#pragma unroll
      for (int hh = 0; hh < 16; ++hh)
        p[hh] = __expf(sc[hh * 512 + ii * 32 + jj] - st_m[hh * 16 + ii]) *
                st_i[hh * 16 + ii];
      float aw[16]; float mu = 0.f, s2 = 0.f;
#pragma unroll
      for (int g = 0; g < 16; ++g) {
        float a = 0.f;
#pragma unroll
        for (int hh = 0; hh < 16; ++hh) a += wt[g * 16 + hh] * p[hh];
        aw[g] = a; mu += a; s2 += a * a;
      }
      mu *= (1.f / 16.f);
      s2 = s2 * (1.f / 16.f) - mu * mu;
      float rinv = rsqrtf(s2 + 1e-5f);
#pragma unroll
      for (int g = 0; g < 16; ++g)
        awb[g * 512 + ii * 32 + jj] =
            f2bf((aw[g] - mu) * rinv * g_sh[g] + b_sh[g]);
    }
    __syncthreads();

    // PV: head h: aw[h] (16x32) @ v_h (32x64); B-frags straight from global vT.
    Frag16 afr;
    {
      const unsigned short* abase = awb + h * 512 + arow * 32 + kb_a;
      U4 lo, hi;
      lo.q = *(const uint4*)(abase);
      hi.q = *(const uint4*)(abase + 16);
#pragma unroll
      for (int e = 0; e < 4; ++e) { afr.u[e] = lo.ui[e]; afr.u[4 + e] = hi.ui[e]; }
    }
    const int kb2 = (lane >> 4) * 8;
#pragma unroll
    for (int c = 0; c < 4; ++c) {
      Frag16 bfr;
      const unsigned short* bbase =
          vtb + (size_t)b * 1024 * 2048 +
          (size_t)(h * 64 + c * 16 + arow) * 2048 + j0 + kb2;
      U4 lo, hi;
      lo.q = *(const uint4*)(bbase);
      hi.q = *(const uint4*)(bbase + 16);
#pragma unroll
      for (int e = 0; e < 4; ++e) { bfr.u[e] = lo.ui[e]; bfr.u[4 + e] = hi.ui[e]; }
      acc[c] = wmma_bf16(afr, bfr, acc[c]);
    }
  }

  // Store: out[b, i, h*64 + d], f32
  const int erow = (lane >> 4) * 8, ecol = lane & 15;
#pragma unroll
  for (int c = 0; c < 4; ++c)
#pragma unroll
    for (int rr = 0; rr < 8; ++rr) {
      int m = i0 + rr + erow;
      int n = h * 64 + c * 16 + ecol;
      out[(size_t)m * 1024 + n] = acc[c][rr];
    }
}

// ---------------------------------------------------------------------------
// Launcher
// ---------------------------------------------------------------------------
extern "C" void kernel_launch(void* const* d_in, const int* in_sizes, int n_in,
                              void* d_out, int out_size, void* d_ws, size_t ws_size,
                              hipStream_t stream) {
  const float* x    = (const float*)d_in[0];
  const float* ctx  = (const float*)d_in[1];
  const float* Wq   = (const float*)d_in[2];
  const float* Wkv  = (const float*)d_in[3];
  const float* Wt   = (const float*)d_in[4];
  const float* gam  = (const float*)d_in[5];
  const float* bet  = (const float*)d_in[6];
  float* out = (float*)d_out;

  // workspace: q (4096x1024), k (4096x1024), vT (2 x 1024x2048) -- bf16, 24MB
  unsigned short* qb  = (unsigned short*)d_ws;
  unsigned short* kbf = qb  + (size_t)4096 * 1024;
  unsigned short* vtb = kbf + (size_t)4096 * 1024;

  // q = x @ Wq^T (scale 1/sqrt(64) folded)
  gemm_xWT_bf16<0><<<dim3(32 * 8), 256, 0, stream>>>(x, Wq, qb, 4096, 1024, 0.125f);
  // k = ctx @ Wkv[:1024]^T
  gemm_xWT_bf16<0><<<dim3(32 * 8), 256, 0, stream>>>(ctx, Wkv, kbf, 4096, 1024, 1.0f);
  // vT = (ctx @ Wkv[1024:]^T)^T, written transposed for WMMA B-operand access
  gemm_xWT_bf16<1><<<dim3(32 * 8), 256, 0, stream>>>(
      ctx, Wkv + (size_t)1024 * 1024, vtb, 4096, 1024, 1.0f);
  // fused talking-heads attention: one block per 16 query rows
  fused_talking_attn<<<dim3(256), 512, LDS_SIZE, stream>>>(
      qb, kbf, vtb, Wt, gam, bet, out);
}
